// FP_86655260164508
// MI455X (gfx1250) — matrix operand order repaired and checked by reading, not compiled, not run on hardware
//
#include <hip/hip_runtime.h>

#define B_SZ   4
#define N_FINE 16384
#define S_CRS  4096
#define CF     128
#define CC     256
#define IN_C   384
#define MID_C  192
#define OUT_C  128
#define NG     8
#define KC     32          // K-chunk per bf16 WMMA
#define LDP    40          // padded LDS row stride (bf16 elems): conflict-free b128 frags

typedef __attribute__((ext_vector_type(8)))  float  v8f;
typedef __attribute__((ext_vector_type(8)))  __bf16 v8bf;
typedef __attribute__((ext_vector_type(16))) __bf16 v16bf;

// ---- fp32 -> bf16 split helpers (RNE, bit math; no __bf16 scalar conv needed) ----
__device__ __forceinline__ unsigned short f2bf(float x) {
  unsigned u = __builtin_bit_cast(unsigned, x);
  unsigned r = (u + 0x7FFFu + ((u >> 16) & 1u)) >> 16;
  return (unsigned short)r;
}
__device__ __forceinline__ float bf2f(unsigned short b) {
  unsigned u = ((unsigned)b) << 16;
  return __builtin_bit_cast(float, u);
}
__device__ __forceinline__ v8bf ld8(const unsigned short* p) {
  return *reinterpret_cast<const v8bf*>(p);
}
__device__ __forceinline__ v16bf cat16(v8bf a, v8bf b) {
  return __builtin_shufflevector(a, b, 0,1,2,3,4,5,6,7,8,9,10,11,12,13,14,15);
}

// ---------------- prep: split weights to bf16 hi/lo + zero stats ----------------
__global__ __launch_bounds__(256) void prep_kernel(
    const float* __restrict__ w1, const float* __restrict__ w2,
    unsigned short* __restrict__ w1h, unsigned short* __restrict__ w1l,
    unsigned short* __restrict__ w2h, unsigned short* __restrict__ w2l,
    float* __restrict__ stats) {
  int i = blockIdx.x * blockDim.x + threadIdx.x;
  int stride = gridDim.x * blockDim.x;
  if (i < 2 * B_SZ * NG * 2) stats[i] = 0.0f;                // 128 floats
  for (int e = i; e < MID_C * IN_C; e += stride) {           // w1: [m][k] row-major
    float x = w1[e];
    unsigned short hi = f2bf(x);
    w1h[e] = hi;
    w1l[e] = f2bf(x - bf2f(hi));
  }
  for (int e = i; e < OUT_C * MID_C; e += stride) {          // w2: [o][k] row-major
    float x = w2[e];
    unsigned short hi = f2bf(x);
    w2h[e] = hi;
    w2l[e] = f2bf(x - bf2f(hi));
  }
}

// ---------------- kNN (K=3) with LDS-resident coarse cloud ----------------
__global__ __launch_bounds__(256) void knn_kernel(
    const float* __restrict__ fine_xyz, const float* __restrict__ coarse_xyz,
    int* __restrict__ idx_out, float* __restrict__ w_out) {
  __shared__ float s_cx[S_CRS * 3];                          // 48 KB
  const int bb = blockIdx.x >> 6;
  const int n  = ((blockIdx.x & 63) << 8) + threadIdx.x;
  for (int i = threadIdx.x; i < S_CRS * 3; i += 256)
    s_cx[i] = coarse_xyz[(size_t)bb * S_CRS * 3 + i];
  __syncthreads();
  const size_t row = (size_t)bb * N_FINE + n;
  const float px = fine_xyz[row * 3 + 0];
  const float py = fine_xyz[row * 3 + 1];
  const float pz = fine_xyz[row * 3 + 2];
  float d0 = 3.4e38f, d1 = 3.4e38f, d2 = 3.4e38f;
  int i0 = 0, i1 = 0, i2 = 0;
  for (int s = 0; s < S_CRS; ++s) {                          // uniform LDS broadcast
    float dx = px - s_cx[3 * s + 0];
    float dy = py - s_cx[3 * s + 1];
    float dz = pz - s_cx[3 * s + 2];
    float d = dx * dx + dy * dy + dz * dz;
    if (d < d0)      { d2 = d1; i2 = i1; d1 = d0; i1 = i0; d0 = d; i0 = s; }
    else if (d < d1) { d2 = d1; i2 = i1; d1 = d;  i1 = s; }
    else if (d < d2) { d2 = d;  i2 = s; }
  }
  float w0 = 1.0f / fmaxf(d0, 1e-8f);
  float w1 = 1.0f / fmaxf(d1, 1e-8f);
  float w2 = 1.0f / fmaxf(d2, 1e-8f);
  float ws = w0 + w1 + w2;
  w0 /= ws; w1 /= ws; w2 /= ws;
  idx_out[row * 3 + 0] = i0; idx_out[row * 3 + 1] = i1; idx_out[row * 3 + 2] = i2;
  w_out[row * 3 + 0] = w0;   w_out[row * 3 + 1] = w1;   w_out[row * 3 + 2] = w2;
}

// ---------------- interpolation + concat (coalesced over channels) ----------------
__global__ __launch_bounds__(128) void interp_kernel(
    const float* __restrict__ fine_feat, const float* __restrict__ coarse_feat,
    const int* __restrict__ idx, const float* __restrict__ w,
    float* __restrict__ combined) {
  const size_t row = blockIdx.x;
  const int bb = (int)(row >> 14);
  const int c  = threadIdx.x;                                // 0..127
  const int   i0 = idx[row * 3], i1 = idx[row * 3 + 1], i2 = idx[row * 3 + 2];
  const float w0 = w[row * 3],   w1 = w[row * 3 + 1],   w2 = w[row * 3 + 2];
  const float* cf = coarse_feat + (size_t)bb * S_CRS * CC;
  combined[row * IN_C + c] = fine_feat[row * CF + c];
  float a0 = w0 * cf[(size_t)i0 * CC + c] + w1 * cf[(size_t)i1 * CC + c]
           + w2 * cf[(size_t)i2 * CC + c];
  float a1 = w0 * cf[(size_t)i0 * CC + 128 + c] + w1 * cf[(size_t)i1 * CC + 128 + c]
           + w2 * cf[(size_t)i2 * CC + 128 + c];
  combined[row * IN_C + CF + c]       = a0;
  combined[row * IN_C + CF + 128 + c] = a1;
}

// ---------------- GEMM1: split-bf16 WMMA, h1 = combined @ w1^T, + GN1 stats ----------------
__global__ __launch_bounds__(256) void gemm1_kernel(
    const float* __restrict__ A,
    const unsigned short* __restrict__ Bh, const unsigned short* __restrict__ Bl,
    float* __restrict__ H, float* __restrict__ stats1) {
  __shared__ alignas(16) unsigned short sAh[128][LDP];
  __shared__ alignas(16) unsigned short sAl[128][LDP];
  __shared__ alignas(16) unsigned short sBh[MID_C][LDP];
  __shared__ alignas(16) unsigned short sBl[MID_C][LDP];
  __shared__ float sStat[2 * NG];
  const int tid  = threadIdx.x;
  const int wave = tid >> 5, lane = tid & 31, half = lane >> 4, lm = lane & 15;
  const int rowBase = blockIdx.x * 128;
  if (tid < 2 * NG) sStat[tid] = 0.0f;
  v8f acc[12] = {};
  for (int kc = 0; kc < IN_C; kc += KC) {
    __syncthreads();
    for (int e = tid; e < 128 * KC; e += 256) {              // stage+split A chunk
      int r = e >> 5, kk = e & 31;
      const float* ap = &A[(size_t)(rowBase + r) * IN_C + kc + kk];
      float x = *ap;
      if (kc + KC < IN_C) __builtin_prefetch(ap + KC, 0, 3); // next K-chunk
      unsigned short hi = f2bf(x);
      sAh[r][kk] = hi;
      sAl[r][kk] = f2bf(x - bf2f(hi));
    }
    for (int e = tid; e < MID_C * KC; e += 256) {            // stage B chunk (pre-split)
      int m = e >> 5, kk = e & 31;
      sBh[m][kk] = Bh[(size_t)m * IN_C + kc + kk];
      sBl[m][kk] = Bl[(size_t)m * IN_C + kc + kk];
    }
    __syncthreads();
    const int arow = wave * 16 + lm;
    v16bf ah = cat16(ld8(&sAh[arow][8 * half]), ld8(&sAh[arow][16 + 8 * half]));
    v16bf al = cat16(ld8(&sAl[arow][8 * half]), ld8(&sAl[arow][16 + 8 * half]));
#pragma unroll
    for (int t = 0; t < 12; ++t) {
      const int col = t * 16 + lm;
      v16bf bh = cat16(ld8(&sBh[col][8 * half]), ld8(&sBh[col][16 + 8 * half]));
      v16bf bl = cat16(ld8(&sBl[col][8 * half]), ld8(&sBl[col][16 + 8 * half]));
      acc[t] = __builtin_amdgcn_wmma_f32_16x16x32_bf16(false, al, false, bh, (short)0, acc[t], false, false);
      acc[t] = __builtin_amdgcn_wmma_f32_16x16x32_bf16(false, ah, false, bl, (short)0, acc[t], false, false);
      acc[t] = __builtin_amdgcn_wmma_f32_16x16x32_bf16(false, ah, false, bh, (short)0, acc[t], false, false);
    }
  }
#pragma unroll
  for (int t = 0; t < 12; ++t) {
    const int col = t * 16 + lm;
    const int g = col / (MID_C / NG);
    float s = 0.0f, s2 = 0.0f;
#pragma unroll
    for (int r = 0; r < 8; ++r) {
      float v = acc[t][r];
      H[(size_t)(rowBase + wave * 16 + (half << 3) + r) * MID_C + col] = v;
      s += v; s2 += v * v;
    }
    atomicAdd(&sStat[g * 2 + 0], s);
    atomicAdd(&sStat[g * 2 + 1], s2);
  }
  __syncthreads();
  if (tid < 2 * NG) {
    const int bb = rowBase >> 14;
    atomicAdd(&stats1[bb * 2 * NG + tid], sStat[tid]);
  }
}

// ---------------- GEMM2: fused GN1+ReLU on A-load, split-bf16 WMMA, + GN2 stats ----------------
__global__ __launch_bounds__(256) void gemm2_kernel(
    const float* __restrict__ H,
    const unsigned short* __restrict__ Bh, const unsigned short* __restrict__ Bl,
    const float* __restrict__ stats1,
    const float* __restrict__ gamma1, const float* __restrict__ beta1,
    float* __restrict__ OUT, float* __restrict__ stats2) {
  __shared__ alignas(16) unsigned short sAh[128][LDP];
  __shared__ alignas(16) unsigned short sAl[128][LDP];
  __shared__ alignas(16) unsigned short sBh[OUT_C][LDP];
  __shared__ alignas(16) unsigned short sBl[OUT_C][LDP];
  __shared__ float sStat[2 * NG];
  __shared__ float sMu[NG], sRs[NG];
  __shared__ float sG[MID_C], sBt[MID_C];
  const int tid  = threadIdx.x;
  const int wave = tid >> 5, lane = tid & 31, half = lane >> 4, lm = lane & 15;
  const int rowBase = blockIdx.x * 128;
  const int bb = rowBase >> 14;
  if (tid < 2 * NG) sStat[tid] = 0.0f;
  if (tid < NG) {
    const float cnt = (float)N_FINE * (float)(MID_C / NG);
    float su = stats1[bb * 2 * NG + tid * 2 + 0];
    float sq = stats1[bb * 2 * NG + tid * 2 + 1];
    float mu = su / cnt;
    float var = sq / cnt - mu * mu;
    sMu[tid] = mu;
    sRs[tid] = rsqrtf(var + 1e-5f);
  }
  for (int i = tid; i < MID_C; i += 256) { sG[i] = gamma1[i]; sBt[i] = beta1[i]; }
  v8f acc[8] = {};
  for (int kc = 0; kc < MID_C; kc += KC) {
    __syncthreads();
    for (int e = tid; e < 128 * KC; e += 256) {              // stage A: GN1+ReLU then split
      int r = e >> 5, kk = e & 31;
      int c = kc + kk;
      float x = H[(size_t)(rowBase + r) * MID_C + c];
      int g = c / (MID_C / NG);
      x = (x - sMu[g]) * sRs[g] * sG[c] + sBt[c];
      x = fmaxf(x, 0.0f);
      unsigned short hi = f2bf(x);
      sAh[r][kk] = hi;
      sAl[r][kk] = f2bf(x - bf2f(hi));
    }
    for (int e = tid; e < OUT_C * KC; e += 256) {            // stage B chunk (pre-split)
      int o = e >> 5, kk = e & 31;
      sBh[o][kk] = Bh[(size_t)o * MID_C + kc + kk];
      sBl[o][kk] = Bl[(size_t)o * MID_C + kc + kk];
    }
    __syncthreads();
    const int arow = wave * 16 + lm;
    v16bf ah = cat16(ld8(&sAh[arow][8 * half]), ld8(&sAh[arow][16 + 8 * half]));
    v16bf al = cat16(ld8(&sAl[arow][8 * half]), ld8(&sAl[arow][16 + 8 * half]));
#pragma unroll
    for (int t = 0; t < 8; ++t) {
      const int col = t * 16 + lm;
      v16bf bh = cat16(ld8(&sBh[col][8 * half]), ld8(&sBh[col][16 + 8 * half]));
      v16bf bl = cat16(ld8(&sBl[col][8 * half]), ld8(&sBl[col][16 + 8 * half]));
      acc[t] = __builtin_amdgcn_wmma_f32_16x16x32_bf16(false, al, false, bh, (short)0, acc[t], false, false);
      acc[t] = __builtin_amdgcn_wmma_f32_16x16x32_bf16(false, ah, false, bl, (short)0, acc[t], false, false);
      acc[t] = __builtin_amdgcn_wmma_f32_16x16x32_bf16(false, ah, false, bh, (short)0, acc[t], false, false);
    }
  }
#pragma unroll
  for (int t = 0; t < 8; ++t) {                              // group == tile (OUT_C/NG == 16)
    const int col = t * 16 + lm;
    float s = 0.0f, s2 = 0.0f;
#pragma unroll
    for (int r = 0; r < 8; ++r) {
      float v = acc[t][r];
      OUT[(size_t)(rowBase + wave * 16 + (half << 3) + r) * OUT_C + col] = v;
      s += v; s2 += v * v;
    }
    atomicAdd(&sStat[t * 2 + 0], s);
    atomicAdd(&sStat[t * 2 + 1], s2);
  }
  __syncthreads();
  if (tid < 2 * NG) atomicAdd(&stats2[bb * 2 * NG + tid], sStat[tid]);
}

// ---------------- GN2 + ReLU in place on d_out ----------------
__global__ __launch_bounds__(256) void gn2_kernel(
    float* __restrict__ OUT, const float* __restrict__ stats2,
    const float* __restrict__ gamma2, const float* __restrict__ beta2) {
  size_t i = (size_t)blockIdx.x * 256 + threadIdx.x;
  if (i >= (size_t)B_SZ * N_FINE * OUT_C) return;
  const int c  = (int)(i & (OUT_C - 1));
  const int bb = (int)(i >> 21);
  const int g  = c >> 4;
  const float cnt = (float)N_FINE * (float)(OUT_C / NG);
  float su = stats2[bb * 2 * NG + g * 2 + 0];
  float sq = stats2[bb * 2 * NG + g * 2 + 1];
  float mu = su / cnt;
  float var = sq / cnt - mu * mu;
  float x = OUT[i];
  x = (x - mu) * rsqrtf(var + 1e-5f) * gamma2[c] + beta2[c];
  OUT[i] = fmaxf(x, 0.0f);
}

extern "C" void kernel_launch(void* const* d_in, const int* in_sizes, int n_in,
                              void* d_out, int out_size, void* d_ws, size_t ws_size,
                              hipStream_t stream) {
  const float* fine_xyz    = (const float*)d_in[0];
  const float* coarse_xyz  = (const float*)d_in[1];
  const float* fine_feat   = (const float*)d_in[2];
  const float* coarse_feat = (const float*)d_in[3];
  const float* w1  = (const float*)d_in[4];
  const float* g1w = (const float*)d_in[5];
  const float* g1b = (const float*)d_in[6];
  const float* w2  = (const float*)d_in[7];
  const float* g2w = (const float*)d_in[8];
  const float* g2b = (const float*)d_in[9];
  float* out = (float*)d_out;

  const size_t ROWS = (size_t)B_SZ * N_FINE;                 // 65536
  float* combined     = (float*)d_ws;                        // ROWS*IN_C f32
  float* h1           = combined + ROWS * IN_C;              // ROWS*MID_C f32
  unsigned short* w1h = (unsigned short*)(h1 + ROWS * MID_C);// 192*384 bf16 each
  unsigned short* w1l = w1h + (size_t)MID_C * IN_C;
  unsigned short* w2h = w1l + (size_t)MID_C * IN_C;          // 128*192 bf16 each
  unsigned short* w2l = w2h + (size_t)OUT_C * MID_C;
  float* wbuf         = (float*)(w2l + (size_t)OUT_C * MID_C); // ROWS*3
  int*   ibuf         = (int*)(wbuf + ROWS * 3);             // ROWS*3
  float* stats        = (float*)(ibuf + ROWS * 3);           // 128 floats
  float* stats1       = stats;
  float* stats2       = stats + B_SZ * NG * 2;

  prep_kernel  <<<288, 256, 0, stream>>>(w1, w2, w1h, w1l, w2h, w2l, stats);
  knn_kernel   <<<B_SZ * 64, 256, 0, stream>>>(fine_xyz, coarse_xyz, ibuf, wbuf);
  interp_kernel<<<(int)ROWS, 128, 0, stream>>>(fine_feat, coarse_feat, ibuf, wbuf, combined);
  gemm1_kernel <<<(int)(ROWS / 128), 256, 0, stream>>>(combined, w1h, w1l, h1, stats1);
  gemm2_kernel <<<(int)(ROWS / 128), 256, 0, stream>>>(h1, w2h, w2l, stats1, g1w, g1b, out, stats2);
  gn2_kernel   <<<(int)((ROWS * OUT_C + 255) / 256), 256, 0, stream>>>(out, stats2, g2w, g2b);
}
